// FreeFlowFPN_17128329577031
// MI455X (gfx1250) — compile-verified, matrix operand order checked by reading, not run on hardware
//
#include <hip/hip_runtime.h>
#include <hip/hip_bf16.h>

typedef __attribute__((ext_vector_type(16))) __bf16 v16bf;
typedef __attribute__((ext_vector_type(8)))  float  v8f;

typedef int v4i_ __attribute__((__vector_size__(16)));
typedef __attribute__((address_space(1))) v4i_ as1_v4i;
typedef __attribute__((address_space(3))) v4i_ as3_v4i;

union FragU { v16bf v; uint4 q[2]; };

__device__ __forceinline__ unsigned short f2bf(float f) {
  unsigned int u = __float_as_uint(f);
  u += 0x7FFFu + ((u >> 16) & 1u);     // round-to-nearest-even (inputs finite)
  return (unsigned short)(u >> 16);
}

#define CCH 256
#define APITCH 264   // 32ch*8chunks + 8 pad: 16B-aligned rows, conflict-free b128

#if defined(__HIP_DEVICE_COMPILE__) && __has_builtin(__builtin_amdgcn_global_load_async_to_lds_b128)
#define HAVE_ASYNC 1
#else
#define HAVE_ASYNC 0
#endif

#if HAVE_ASYNC
#if __has_builtin(__builtin_amdgcn_s_wait_asynccnt)
#define ASYNC_WAIT() __builtin_amdgcn_s_wait_asynccnt(0)
#else
#define ASYNC_WAIT() asm volatile("s_wait_asynccnt 0x0" ::: "memory")
#endif
#else
#define ASYNC_WAIT()
#endif

// ---------------------------------------------------------------------------
// 1) NCHW -> [HW][C] fp32 transpose (per level)
// ---------------------------------------------------------------------------
__global__ __launch_bounds__(256)
void nchw_to_pc(const float* __restrict__ src, float* __restrict__ dst, int HW) {
  const int c = threadIdx.x;
  const int pbase = blockIdx.x * 16;
#pragma unroll
  for (int pp = 0; pp < 16; ++pp) {
    const int p = pbase + pp;
    dst[(size_t)p * CCH + c] = src[(size_t)c * HW + p];
  }
}

// ---------------------------------------------------------------------------
// 2) Global average pool: block = (level, channel)
// ---------------------------------------------------------------------------
__global__ __launch_bounds__(256)
void gap_kernel(const float* __restrict__ f0, const float* __restrict__ f1,
                const float* __restrict__ f2, const float* __restrict__ f3,
                float* __restrict__ vecs) {
  __shared__ float red[256];
  const int l = blockIdx.x >> 8;
  const int c = blockIdx.x & 255;
  const float* srcs[4] = {f0, f1, f2, f3};
  const int hws[4] = {65536, 16384, 4096, 1024};
  const int HW = hws[l];
  const float* base = srcs[l] + (size_t)c * HW;
  float s = 0.f;
  for (int p = threadIdx.x; p < HW; p += 256) s += base[p];
  red[threadIdx.x] = s;
  __syncthreads();
  for (int off = 128; off > 0; off >>= 1) {
    if (threadIdx.x < off) red[threadIdx.x] += red[threadIdx.x + off];
    __syncthreads();
  }
  if (threadIdx.x == 0) vecs[l * CCH + c] = red[0] * (1.0f / (float)HW);
}

// ---------------------------------------------------------------------------
// 3) Scoring MLP, argsort order, q/k attention, softmax, TAU threshold
// ---------------------------------------------------------------------------
__global__ __launch_bounds__(256)
void score_attn_kernel(const float* __restrict__ vecs,
                       const float* __restrict__ mlp_w1, const float* __restrict__ mlp_b1,
                       const float* __restrict__ mlp_w2, const float* __restrict__ mlp_b2,
                       const float* __restrict__ qk_w,  const float* __restrict__ qk_b,
                       int* __restrict__ order_out, float* __restrict__ w_out) {
  __shared__ float vs[4 * 256];
  __shared__ float hs[4 * 64];
  __shared__ float qs[4 * 256];
  __shared__ float ksm[4 * 256];
  __shared__ float sc[4];
  __shared__ float at[16];
  const int t = threadIdx.x;
  for (int i = t; i < 1024; i += 256) vs[i] = vecs[i];
  __syncthreads();
  {
    const int l = t >> 6, j = t & 63;
    float s = mlp_b1[j];
    for (int c = 0; c < 256; ++c) s += vs[l * 256 + c] * mlp_w1[c * 64 + j];
    hs[t] = s > 0.f ? s : 0.f;
  }
  __syncthreads();
  if (t < 4) {
    float s = mlp_b2[0];
    for (int j = 0; j < 64; ++j) s += hs[t * 64 + j] * mlp_w2[j];
    sc[t] = s;
  }
  __syncthreads();
  for (int idx = t; idx < 2048; idx += 256) {
    const int l = idx >> 9, d = idx & 511;
    float s = qk_b[d];
    for (int c = 0; c < 256; ++c) s += vs[l * 256 + c] * qk_w[c * 512 + d];
    if (d < 256) qs[l * 256 + d] = s;
    else         ksm[l * 256 + (d - 256)] = s;
  }
  __syncthreads();
  if (t < 16) {
    const int i = t >> 2, j = t & 3;
    float aij;
    if (i == j) aij = -__builtin_inff();
    else {
      float s = 0.f;
      for (int c = 0; c < 256; ++c) s += qs[i * 256 + c] * ksm[j * 256 + c];
      aij = s * (1.0f / 16.0f);  // / sqrt(256)
    }
    at[t] = aij;
  }
  __syncthreads();
  if (t < 4) {
    float m = at[t * 4];
    for (int j = 1; j < 4; ++j) m = fmaxf(m, at[t * 4 + j]);
    float e[4], s = 0.f;
    for (int j = 0; j < 4; ++j) { e[j] = expf(at[t * 4 + j] - m); s += e[j]; }
    for (int j = 0; j < 4; ++j) {
      float w = e[j] / s;
      if (t == j) w = 0.f;
      w_out[t * 4 + j] = (w >= 0.1f) ? w : 0.f;   // TAU pruning
    }
  }
  if (t == 0) {  // stable argsort descending (ties -> lower index first)
    float s2[4];
    for (int l = 0; l < 4; ++l) s2[l] = sc[l];
    for (int r = 0; r < 4; ++r) {
      int best = 0; float bv = s2[0];
      for (int l = 1; l < 4; ++l) if (s2[l] > bv) { bv = s2[l]; best = l; }
      order_out[r] = best;
      s2[best] = -__builtin_inff();
    }
  }
}

// ---------------------------------------------------------------------------
// 4) One fusion step: feats[b] += w * bilinear(feats[a]); (a,b) from device order
// ---------------------------------------------------------------------------
__global__ __launch_bounds__(256)
void fuse_kernel(float* __restrict__ g0, float* __restrict__ g1,
                 float* __restrict__ g2, float* __restrict__ g3,
                 const int* __restrict__ order, const float* __restrict__ wmat,
                 int oi, int oj) {
  const int a = order[oi];
  const int b = order[oj];
  const float w = wmat[a * 4 + b];
  if (w == 0.f) return;
  float* fp[4] = {g0, g1, g2, g3};
  const int Htab[4] = {256, 128, 64, 32};
  const int Hb = Htab[b], Wb = Hb, Ha = Htab[a], Wa = Ha;
  const long total = (long)Hb * Wb * 64;           // 4 channels per thread
  const long idx = (long)blockIdx.x * 256 + threadIdx.x;
  if (idx >= total) return;
  const int c0 = (int)(idx & 63) * 4;
  const int p = (int)(idx >> 6);
  const int lwb = 31 - __clz(Wb);
  const int y = p >> lwb, x = p & (Wb - 1);
  const float ratio = (float)Ha / (float)Hb;
  float sy = fminf(fmaxf(((float)y + 0.5f) * ratio - 0.5f, 0.f), (float)(Ha - 1));
  float sx = fminf(fmaxf(((float)x + 0.5f) * ratio - 0.5f, 0.f), (float)(Wa - 1));
  const int y0 = (int)floorf(sy); const int y1 = min(y0 + 1, Ha - 1); const float wy = sy - (float)y0;
  const int x0 = (int)floorf(sx); const int x1 = min(x0 + 1, Wa - 1); const float wx = sx - (float)x0;
  const float* A = fp[a];
  float* B = fp[b];
  const float* r00 = A + (size_t)(y0 * Wa + x0) * CCH + c0;
  const float* r01 = A + (size_t)(y0 * Wa + x1) * CCH + c0;
  const float* r10 = A + (size_t)(y1 * Wa + x0) * CCH + c0;
  const float* r11 = A + (size_t)(y1 * Wa + x1) * CCH + c0;
  float* dst = B + (size_t)p * CCH + c0;
#pragma unroll
  for (int q = 0; q < 4; ++q) {
    const float s = (1.f - wy) * ((1.f - wx) * r00[q] + wx * r01[q]) +
                    wy * ((1.f - wx) * r10[q] + wx * r11[q]);
    dst[q] += w * s;
  }
}

// ---------------------------------------------------------------------------
// 5) fp32 [HW][C] -> bf16 [HW][C] (post-fusion, feeds WMMA A path)
// ---------------------------------------------------------------------------
__global__ __launch_bounds__(256)
void f32_to_bf16_kernel(const float* __restrict__ src,
                        unsigned short* __restrict__ dst, long n8) {
  const long i = (long)blockIdx.x * 256 + threadIdx.x;
  if (i >= n8) return;
  const float4 a = *(const float4*)(src + i * 8);
  const float4 b = *(const float4*)(src + i * 8 + 4);
  uint4 o;
  o.x = (unsigned int)f2bf(a.x) | ((unsigned int)f2bf(a.y) << 16);
  o.y = (unsigned int)f2bf(a.z) | ((unsigned int)f2bf(a.w) << 16);
  o.z = (unsigned int)f2bf(b.x) | ((unsigned int)f2bf(b.y) << 16);
  o.w = (unsigned int)f2bf(b.z) | ((unsigned int)f2bf(b.w) << 16);
  *(uint4*)(dst + i * 8) = o;
}

// ---------------------------------------------------------------------------
// 6) Pack conv weights OIHW f32 -> bf16 B-fragment order [tap][kc][n][kh][e]
//    e -> K: e<8 ? kh*8+e : kh*8+e+8   (WMMA 16-bit B layout); also zero-page
// ---------------------------------------------------------------------------
__global__ __launch_bounds__(256)
void pack_weights(const float* __restrict__ w0, const float* __restrict__ w1,
                  const float* __restrict__ w2, const float* __restrict__ w3,
                  unsigned short* __restrict__ Bpack,
                  unsigned short* __restrict__ zpage) {
  if (blockIdx.x == 0) zpage[threadIdx.x] = 0;
  const long idx = (long)blockIdx.x * 256 + threadIdx.x;
  if (idx >= 4L * 589824L) return;
  const int lvl = (int)(idx / 589824L);
  const int rem = (int)(idx % 589824L);
  const int e = rem & 15;
  const int kh = (rem >> 4) & 1;
  const int n = (rem >> 5) & 255;
  const int kc = (rem >> 13) & 7;
  const int tap = rem >> 16;
  const int K = kh * 8 + (e < 8 ? e : e + 8);
  const int ci = kc * 32 + K;
  const float* w = (lvl == 0) ? w0 : (lvl == 1) ? w1 : (lvl == 2) ? w2 : w3;
  Bpack[idx] = f2bf(w[(size_t)n * 2304 + ci * 9 + tap]);
}

// ---------------------------------------------------------------------------
// 7) 3x3 SAME conv: implicit GEMM, bf16 WMMA, f32 accum.
//    WG: 32 pixels x 256 out-ch, 8 waves, wave tile 16x64 (4 wmma tiles).
//    A slab (32 px x 256 ch bf16) staged per tap, double-buffered via
//    global_load_async_to_lds_b128 (fallback: plain ld/ds_store).
// ---------------------------------------------------------------------------
#define WMMA_BF16(ACC, AV, BV) \
  ACC = __builtin_amdgcn_wmma_f32_16x16x32_bf16(false, AV, false, BV, (short)0, ACC, false, false)

__device__ __forceinline__ void stage_slab(const unsigned short* __restrict__ featsB,
                                           const unsigned short* __restrict__ zpage,
                                           unsigned short* buf,
                                           int p0, int H, int W, int lw,
                                           int dy, int dx, int t) {
  const int fr = t >> 3;          // slab row 0..31
  const int fc = (t & 7) << 5;    // channel chunk: 0,32,...,224
  const int pf = p0 + fr;
  const int yy = (pf >> lw) + dy;
  const int xx = (pf & (W - 1)) + dx;
  const bool inb = (yy >= 0) && (yy < H) && (xx >= 0) && (xx < W);
  const unsigned short* g = inb ? (featsB + (size_t)(yy * W + xx) * CCH + fc)
                                : (zpage + fc);
  unsigned short* l = buf + fr * APITCH + fc;
#if HAVE_ASYNC
#pragma unroll
  for (int j = 0; j < 4; ++j)
    __builtin_amdgcn_global_load_async_to_lds_b128(
        (as1_v4i*)(g + j * 8), (as3_v4i*)(l + j * 8), 0, 0);
#else
#pragma unroll
  for (int j = 0; j < 4; ++j)
    *(uint4*)(l + j * 8) = *(const uint4*)(g + j * 8);
#endif
}

__global__ __launch_bounds__(256)
void conv3x3_wmma_kernel(const unsigned short* __restrict__ featsB,
                         const unsigned short* __restrict__ Bpack,
                         const unsigned short* __restrict__ zpage,
                         const float* __restrict__ bias,
                         float* __restrict__ out, int H, int W) {
  __shared__ __align__(16) unsigned short As[2][32 * APITCH];
  const int HW = H * W;
  const int t = threadIdx.x;
  const int wave = t >> 5;
  const int lane = t & 31;
  const int lm = lane & 15;
  const int kh = lane >> 4;
  const int m_blk = (wave & 1) << 4;    // 0 / 16
  const int n_base = (wave >> 1) << 6;  // 0 / 64 / 128 / 192
  const int p0 = blockIdx.x << 5;
  const int lw = 31 - __clz(W);

  v8f acc0 = {}, acc1 = {}, acc2 = {}, acc3 = {};

  stage_slab(featsB, zpage, As[0], p0, H, W, lw, -1, -1, t);
  ASYNC_WAIT();
  __syncthreads();

  for (int tap = 0; tap < 9; ++tap) {
    const unsigned short* cur = As[tap & 1];
    if (tap < 8) {
      const int nx = tap + 1;
      stage_slab(featsB, zpage, As[nx & 1], p0, H, W, lw, nx / 3 - 1, nx % 3 - 1, t);
    }
    const unsigned short* bt = Bpack + ((size_t)tap << 16) + ((size_t)kh << 4);
    const unsigned short* arow = cur + (m_blk + lm) * APITCH + (kh << 3);

#pragma unroll
    for (int kc = 0; kc < 8; ++kc) {
      // A fragment: two contiguous 16B groups (K = kh*8.. and kh*8+16..)
      FragU a;
      const unsigned short* ar = arow + (kc << 5);
      a.q[0] = *(const uint4*)(ar);
      a.q[1] = *(const uint4*)(ar + 16);
      // B fragments straight from global (L2-resident packed weights)
      const unsigned short* bb = bt + ((size_t)kc << 13);
      const unsigned short* q0 = bb + ((size_t)(n_base + 0 + lm) << 5);
      const unsigned short* q1 = bb + ((size_t)(n_base + 16 + lm) << 5);
      const unsigned short* q2 = bb + ((size_t)(n_base + 32 + lm) << 5);
      const unsigned short* q3 = bb + ((size_t)(n_base + 48 + lm) << 5);
      FragU b0, b1, b2, b3;
      b0.q[0] = *(const uint4*)(q0); b0.q[1] = *(const uint4*)(q0 + 8);
      b1.q[0] = *(const uint4*)(q1); b1.q[1] = *(const uint4*)(q1 + 8);
      b2.q[0] = *(const uint4*)(q2); b2.q[1] = *(const uint4*)(q2 + 8);
      b3.q[0] = *(const uint4*)(q3); b3.q[1] = *(const uint4*)(q3 + 8);
      WMMA_BF16(acc0, a.v, b0.v);
      WMMA_BF16(acc1, a.v, b1.v);
      WMMA_BF16(acc2, a.v, b2.v);
      WMMA_BF16(acc3, a.v, b3.v);
    }
    ASYNC_WAIT();      // next-tap slab complete
    __syncthreads();   // all waves done reading cur
  }

  // Epilogue: C/D layout -> lane = N, VGPR r = M (+8 for lanes 16-31)
  const int pe = p0 + m_blk + (kh << 3);
  {
    const int n = n_base + lm; const float bv = bias[n];
    float* o = out + (size_t)n * HW + pe;
#pragma unroll
    for (int r = 0; r < 8; ++r) o[r] = fmaxf(acc0[r] + bv, 0.f);
  }
  {
    const int n = n_base + 16 + lm; const float bv = bias[n];
    float* o = out + (size_t)n * HW + pe;
#pragma unroll
    for (int r = 0; r < 8; ++r) o[r] = fmaxf(acc1[r] + bv, 0.f);
  }
  {
    const int n = n_base + 32 + lm; const float bv = bias[n];
    float* o = out + (size_t)n * HW + pe;
#pragma unroll
    for (int r = 0; r < 8; ++r) o[r] = fmaxf(acc2[r] + bv, 0.f);
  }
  {
    const int n = n_base + 48 + lm; const float bv = bias[n];
    float* o = out + (size_t)n * HW + pe;
#pragma unroll
    for (int r = 0; r < 8; ++r) o[r] = fmaxf(acc3[r] + bv, 0.f);
  }
}

// ---------------------------------------------------------------------------
extern "C" void kernel_launch(void* const* d_in, const int* in_sizes, int n_in,
                              void* d_out, int out_size, void* d_ws, size_t ws_size,
                              hipStream_t stream) {
  const float* f[4] = {(const float*)d_in[0], (const float*)d_in[1],
                       (const float*)d_in[2], (const float*)d_in[3]};
  const float* mlp_w1 = (const float*)d_in[4];
  const float* mlp_b1 = (const float*)d_in[5];
  const float* mlp_w2 = (const float*)d_in[6];
  const float* mlp_b2 = (const float*)d_in[7];
  const float* qk_w = (const float*)d_in[8];
  const float* qk_b = (const float*)d_in[9];
  // setup_inputs interleaves conv_w{i}, conv_b{i}
  const float* conv_w[4] = {(const float*)d_in[10], (const float*)d_in[12],
                            (const float*)d_in[14], (const float*)d_in[16]};
  const float* conv_b[4] = {(const float*)d_in[11], (const float*)d_in[13],
                            (const float*)d_in[15], (const float*)d_in[17]};

  char* ws = (char*)d_ws;
  float* featsT[4] = {(float*)(ws + 0),
                      (float*)(ws + 67108864),
                      (float*)(ws + 83886080),
                      (float*)(ws + 88080384)};
  unsigned short* featsB[4] = {(unsigned short*)(ws + 89128960),
                               (unsigned short*)(ws + 122683392),
                               (unsigned short*)(ws + 131072000),
                               (unsigned short*)(ws + 133169152)};
  unsigned short* Bpack = (unsigned short*)(ws + 133693440);
  unsigned short* zpage = (unsigned short*)(ws + 138412032);
  float* vecs = (float*)(ws + 138412544);
  int* order = (int*)(ws + 138416640);
  float* wmat = (float*)(ws + 138416656);
  float* out = (float*)d_out;

  const int Hs[4] = {256, 128, 64, 32};
  const size_t outOff[4] = {0, 16777216, 20971520, 22020096};

  for (int l = 0; l < 4; ++l) {
    const int HW = Hs[l] * Hs[l];
    nchw_to_pc<<<HW / 16, 256, 0, stream>>>(f[l], featsT[l], HW);
  }
  gap_kernel<<<1024, 256, 0, stream>>>(f[0], f[1], f[2], f[3], vecs);
  score_attn_kernel<<<1, 256, 0, stream>>>(vecs, mlp_w1, mlp_b1, mlp_w2, mlp_b2,
                                           qk_w, qk_b, order, wmat);
  pack_weights<<<9216, 256, 0, stream>>>(conv_w[0], conv_w[1], conv_w[2], conv_w[3],
                                         Bpack, zpage);

  // 12 ordered fusion steps (i-major, j-minor, skip diagonal) — matches the
  // reference's sequential, data-dependent update chain.
  for (int i = 0; i < 4; ++i)
    for (int j = 0; j < 4; ++j)
      if (i != j)
        fuse_kernel<<<16384, 256, 0, stream>>>(featsT[0], featsT[1], featsT[2], featsT[3],
                                               order, wmat, i, j);

  for (int l = 0; l < 4; ++l) {
    const long n8 = (long)Hs[l] * Hs[l] * 32;  // 8 floats per thread
    f32_to_bf16_kernel<<<(int)((n8 + 255) / 256), 256, 0, stream>>>(featsT[l], featsB[l], n8);
  }

  for (int l = 0; l < 4; ++l) {
    const int HW = Hs[l] * Hs[l];
    conv3x3_wmma_kernel<<<HW / 32, 256, 0, stream>>>(
        featsB[l], Bpack + (size_t)l * 589824, zpage, conv_b[l], out + outOff[l], Hs[l], Hs[l]);
  }
}
